// DMuCA_19447611916374
// MI455X (gfx1250) — matrix operand order repaired and checked by reading, not compiled, not run on hardware
//
#include <hip/hip_runtime.h>
#include <hip/hip_bf16.h>

#define BS   256
#define CHN  512
#define HW   81
#define DD   9
#define SPA  64
#define CHW  (CHN * HW)
#define EPSBN 1e-5f

typedef __attribute__((ext_vector_type(16))) __bf16 bf16x16;
typedef __attribute__((ext_vector_type(8)))  __bf16 bf16x8;
typedef __attribute__((ext_vector_type(8)))  float  f32x8;

__device__ __forceinline__ __bf16 f2bf(float f) {
    unsigned u = __builtin_bit_cast(unsigned, f);
    unsigned r = u + 0x7FFFu + ((u >> 16) & 1u);   // round-to-nearest-even
    unsigned short h = (unsigned short)(r >> 16);
    return __builtin_bit_cast(__bf16, h);
}

// inference BN: p = [gamma(n); beta(n); mean(n); var(n)]
__device__ __forceinline__ float bnf(float x, const float* __restrict__ p, int n, int i) {
    float g = p[i], b = p[n + i], m = p[2 * n + i], va = p[3 * n + i];
    float s = g * rsqrtf(va + EPSBN);
    return (x - m) * s + b;
}

// ---------------------------------------------------------------------------
// K1: fused key/value convs.  One block per (b, c) plane, 81 active lanes.
//   k1  = spectral depthwise 9-tap conv along channel axis (per-pixel filter)
//   v   = spatial depthwise 3x3 + bias
//   k1s = relu(bn(spatial depthwise 3x3))
// ---------------------------------------------------------------------------
__global__ void __launch_bounds__(96)
k_keys(const float* __restrict__ x,
       const float* __restrict__ se_key_w, const float* __restrict__ se_key_b,
       const float* __restrict__ se_val_w, const float* __restrict__ se_val_b,
       const float* __restrict__ sa_key_w, const float* __restrict__ sa_key_bn,
       float* __restrict__ k1, float* __restrict__ v, float* __restrict__ k1s) {
    const int c = blockIdx.x, b = blockIdx.y, tid = threadIdx.x;
    __shared__ float pl[HW];
    const size_t base = (size_t)b * CHW + (size_t)c * HW;
    if (tid < HW) pl[tid] = x[base + tid];
    __syncthreads();
    if (tid >= HW) return;
    const int p = tid, py = p / 9, px = p % 9;

    // spectral depthwise conv (cross-correlation, pad 4) over channel axis
    float s1 = se_key_b[p];
    #pragma unroll
    for (int t = 0; t < DD; ++t) {
        int cc = c + t - 4;
        if (cc >= 0 && cc < CHN)
            s1 += se_key_w[p * DD + t] * x[(size_t)b * CHW + (size_t)cc * HW + p];
    }
    k1[base + p] = s1;

    // spatial 3x3 depthwise (value conv + key conv) from LDS plane
    float sv = 0.f, sk = 0.f;
    #pragma unroll
    for (int dy = 0; dy < 3; ++dy) {
        int y = py + dy - 1;
        if (y < 0 || y >= 9) continue;
        #pragma unroll
        for (int dx = 0; dx < 3; ++dx) {
            int xx = px + dx - 1;
            if (xx < 0 || xx >= 9) continue;
            float in = pl[y * 9 + xx];
            sv += se_val_w[c * 9 + dy * 3 + dx] * in;
            sk += sa_key_w[c * 9 + dy * 3 + dx] * in;
        }
    }
    v[base + p] = sv + se_val_b[c];
    float kk = bnf(sk, sa_key_bn, CHN, c);
    k1s[base + p] = kk > 0.f ? kk : 0.f;
}

// ---------------------------------------------------------------------------
// K2: SeMCA attention.  Per block = (128-col chunk, batch).
// GEMM1 (WMMA bf16, f32 acc): C(96x128) = W1pad(96x192) @ KQ(192x128-chunk)
//   KQ rows 0..80 = x[b] viewed as 81x512, rows 81..161 = k1[b], rest zero.
// Then BN+ReLU -> LDS, W2(9x81) projection + bias, softmax over 9 -> ase.
// ---------------------------------------------------------------------------
__global__ void __launch_bounds__(256)
k_semca_attn(const float* __restrict__ x, const float* __restrict__ k1,
             const float* __restrict__ w1, const float* __restrict__ bn1,
             const float* __restrict__ w2, const float* __restrict__ b2,
             float* __restrict__ ase) {
    const int nb0 = blockIdx.x * 128;
    const int b   = blockIdx.y;
    const int tid = threadIdx.x, lane = tid & 31, wv = tid >> 5;

    __shared__ alignas(32) unsigned char smem[49152];
    __bf16* As = (__bf16*)smem;             // 96 x 192 row-major (pitch 192)
    __bf16* Bs = (__bf16*)(smem + 36864);   // 128 cols x 32 k, col-major (pitch 32)
    float*  Cs = (float*)smem;              // 81(96) x 128 f32, reused after GEMM

    // stage A = W1 zero-padded to 96x192, converted to bf16
    for (int idx = tid; idx < 96 * 192; idx += 256) {
        int m = idx / 192, k = idx % 192;
        float val = (m < 81 && k < 162) ? w1[m * 162 + k] : 0.f;
        As[idx] = f2bf(val);
    }

    f32x8 acc[6] = {};
    const float* xb  = x  + (size_t)b * CHW;
    const float* k1b = k1 + (size_t)b * CHW;

    for (int ks = 0; ks < 6; ++ks) {
        const int k0 = ks * 32;
        __syncthreads();   // A staged (first iter) / previous B consumed
        #pragma unroll
        for (int it = 0; it < 16; ++it) {
            int e  = tid + 256 * it;          // 32 rows x 128 cols
            int r  = e >> 7, cl = e & 127;
            int i  = k0 + r;
            float val = 0.f;
            if (i < 81)       val = xb[i * 512 + nb0 + cl];
            else if (i < 162) val = k1b[(i - 81) * 512 + nb0 + cl];
            Bs[cl * 32 + r] = f2bf(val);
        }
        __syncthreads();

        // B fragment: lane -> N = lane&15, K in {kb..kb+7, 16+kb..16+kb+7}, kb = 8*(lane>=16)
        const __bf16* bp = Bs + (wv * 16 + (lane & 15)) * 32 + 8 * (lane >> 4);
        bf16x8 blo = *(const bf16x8*)bp;
        bf16x8 bhi = *(const bf16x8*)(bp + 16);
        bf16x16 bfrag;
        #pragma unroll
        for (int i = 0; i < 8; ++i) { bfrag[i] = blo[i]; bfrag[8 + i] = bhi[i]; }

        #pragma unroll
        for (int m = 0; m < 6; ++m) {
            // A fragment: lane -> M = lane&15, K in {kb..kb+7, 16+kb..16+kb+7}, kb = 8*(lane>=16)
            const __bf16* ap = As + (m * 16 + (lane & 15)) * 192 + k0 + 8 * (lane >> 4);
            bf16x8 alo = *(const bf16x8*)ap;
            bf16x8 ahi = *(const bf16x8*)(ap + 16);
            bf16x16 afrag;
            #pragma unroll
            for (int i = 0; i < 8; ++i) { afrag[i] = alo[i]; afrag[8 + i] = ahi[i]; }
            acc[m] = __builtin_amdgcn_wmma_f32_16x16x32_bf16(
                false, afrag, false, bfrag, (short)0, acc[m], false, false);
        }
    }
    __syncthreads();   // all A/B LDS reads done; safe to overwrite with Cs

    // scatter accumulators with BN + ReLU.  C layout: row = 16m + e + 8*(lane>=16), col = lane&15
    #pragma unroll
    for (int m = 0; m < 6; ++m) {
        #pragma unroll
        for (int e = 0; e < 8; ++e) {
            int row = m * 16 + e + 8 * (lane >> 4);
            if (row < 81) {
                float val = bnf(acc[m][e], bn1, 81, row);
                Cs[row * 128 + wv * 16 + (lane & 15)] = val > 0.f ? val : 0.f;
            }
        }
    }
    __syncthreads();

    // W2 (9x81) projection + bias, softmax over the 9 spectral taps
    if (tid < 128) {
        float s[DD];
        #pragma unroll
        for (int d = 0; d < DD; ++d) s[d] = b2[d];
        for (int o = 0; o < 81; ++o) {
            float cv = Cs[o * 128 + tid];
            #pragma unroll
            for (int d = 0; d < DD; ++d) s[d] += w2[d * 81 + o] * cv;
        }
        float mx = s[0];
        #pragma unroll
        for (int d = 1; d < DD; ++d) mx = fmaxf(mx, s[d]);
        float sum = 0.f;
        #pragma unroll
        for (int d = 0; d < DD; ++d) { s[d] = __expf(s[d] - mx); sum += s[d]; }
        float inv = 1.f / sum;
        const int gc = nb0 + tid;
        #pragma unroll
        for (int d = 0; d < DD; ++d) ase[((size_t)b * DD + d) * CHN + gc] = s[d] * inv;
    }
}

// ---------------------------------------------------------------------------
// K3: SaMCA attention.  One block per (b, head g); grouped 1x1 convs (16->8->1)
// + BN/ReLU + global spatial softmax over 81 pixels.
// ---------------------------------------------------------------------------
__global__ void __launch_bounds__(128)
k_samca_attn(const float* __restrict__ x, const float* __restrict__ k1s,
             const float* __restrict__ w1, const float* __restrict__ bn1,
             const float* __restrict__ w2, const float* __restrict__ b2,
             float* __restrict__ a2s) {
    const int g = blockIdx.x, b = blockIdx.y, tid = threadIdx.x;
    __shared__ float lx[8 * HW], lk[8 * HW], sl[HW], red;
    for (int idx = tid; idx < 8 * HW; idx += 128) {
        int j = idx / HW, p = idx % HW;
        size_t gi = (size_t)b * CHW + (size_t)(g * 8 + j) * HW + p;
        lx[idx] = x[gi];
        lk[idx] = k1s[gi];
    }
    __syncthreads();
    float logit = 0.f;
    const bool act = tid < HW;
    if (act) {
        logit = b2[g];
        #pragma unroll
        for (int oc = 0; oc < 8; ++oc) {
            int o = g * 8 + oc;
            float s = 0.f;
            #pragma unroll
            for (int j = 0; j < 8; ++j) {       // interleaved [k1s, x] channels
                s += w1[o * 16 + 2 * j]     * lk[j * HW + tid];
                s += w1[o * 16 + 2 * j + 1] * lx[j * HW + tid];
            }
            s = bnf(s, bn1, CHN, o);
            s = s > 0.f ? s : 0.f;
            logit += w2[g * 8 + oc] * s;
        }
        sl[tid] = logit;
    }
    __syncthreads();
    if (tid == 0) {
        float m = sl[0];
        for (int i = 1; i < HW; ++i) m = fmaxf(m, sl[i]);
        red = m;
    }
    __syncthreads();
    float e = 0.f;
    if (act) { e = __expf(logit - red); sl[tid] = e; }
    __syncthreads();
    if (tid == 0) {
        float s = 0.f;
        for (int i = 0; i < HW; ++i) s += sl[i];
        red = s;
    }
    __syncthreads();
    if (act) a2s[((size_t)b * SPA + g) * HW + tid] = e / red;
}

// ---------------------------------------------------------------------------
// K4: final fuse:  out = alpha*(k1 + sum_d ase[d]*v[c+d-4]) + (1-alpha)*(k1s + a2s*x)
// ---------------------------------------------------------------------------
__global__ void __launch_bounds__(96)
k_fuse(const float* __restrict__ x, const float* __restrict__ alpha,
       const float* __restrict__ k1, const float* __restrict__ v,
       const float* __restrict__ k1s, const float* __restrict__ ase,
       const float* __restrict__ a2s, float* __restrict__ out) {
    const int c = blockIdx.x, b = blockIdx.y, tid = threadIdx.x;
    __shared__ float aw[DD];
    if (tid < DD) aw[tid] = ase[((size_t)b * DD + tid) * CHN + c];
    __syncthreads();
    if (tid >= HW) return;
    const size_t base = (size_t)b * CHW + (size_t)c * HW;
    float o1 = k1[base + tid];
    #pragma unroll
    for (int d = 0; d < DD; ++d) {
        int cc = c + d - 4;
        if (cc >= 0 && cc < CHN)
            o1 += aw[d] * v[(size_t)b * CHW + (size_t)cc * HW + tid];
    }
    float a2v = a2s[((size_t)b * SPA + (c >> 3)) * HW + tid];
    float o2 = k1s[base + tid] + a2v * x[base + tid];
    float al = alpha[0];
    out[base + tid] = al * o1 + (1.f - al) * o2;
}

extern "C" void kernel_launch(void* const* d_in, const int* in_sizes, int n_in,
                              void* d_out, int out_size, void* d_ws, size_t ws_size,
                              hipStream_t stream) {
    const float* x         = (const float*)d_in[0];
    const float* alpha     = (const float*)d_in[1];
    const float* sa_key_w  = (const float*)d_in[2];
    const float* sa_key_bn = (const float*)d_in[3];
    const float* sa_att_w1 = (const float*)d_in[4];
    const float* sa_att_bn = (const float*)d_in[5];
    const float* sa_att_w2 = (const float*)d_in[6];
    const float* sa_att_b2 = (const float*)d_in[7];
    const float* se_key_w  = (const float*)d_in[8];
    const float* se_key_b  = (const float*)d_in[9];
    const float* se_att_w1 = (const float*)d_in[10];
    const float* se_att_bn = (const float*)d_in[11];
    const float* se_att_w2 = (const float*)d_in[12];
    const float* se_att_b2 = (const float*)d_in[13];
    const float* se_val_w  = (const float*)d_in[14];
    const float* se_val_b  = (const float*)d_in[15];

    float* ws = (float*)d_ws;
    const size_t N = (size_t)BS * CHW;          // 10,616,832
    float* k1  = ws;                            // (b,c,hw)
    float* v   = ws + N;                        // (b,c,hw)
    float* k1s = ws + 2 * N;                    // (b,c,hw)
    float* ase = ws + 3 * N;                    // (b,9,512)
    float* a2s = ase + (size_t)BS * DD * CHN;   // (b,64,81)

    k_keys<<<dim3(CHN, BS), 96, 0, stream>>>(
        x, se_key_w, se_key_b, se_val_w, se_val_b, sa_key_w, sa_key_bn, k1, v, k1s);
    k_semca_attn<<<dim3(4, BS), 256, 0, stream>>>(
        x, k1, se_att_w1, se_att_bn, se_att_w2, se_att_b2, ase);
    k_samca_attn<<<dim3(SPA, BS), 128, 0, stream>>>(
        x, k1s, sa_att_w1, sa_att_bn, sa_att_w2, sa_att_b2, a2s);
    k_fuse<<<dim3(CHN, BS), 96, 0, stream>>>(
        x, alpha, k1, v, k1s, ase, a2s, (float*)d_out);
}
